// MaskedBlockLinear_63909113364673
// MI455X (gfx1250) — matrix operand order repaired and checked by reading, not compiled, not run on hardware
//
#include <hip/hip_runtime.h>
#include <hip/hip_bf16.h>

// ---------------------------------------------------------------------------
// MaskedBlockLinear on gfx1250 (MI455X).
// Pass 1: split f32 X and the block-diagonal slice of W into bf16 hi/lo
//         planes once (removes per-tile conversion VALU entirely).
// Pass 2: 8 independent [8192 x 1024 x 256] GEMMs with
//         v_wmma_f32_16x16x32_bf16, 3-term (hh + hl + lh) f32 emulation.
//         Wave tile 32x64 (2x4 WMMA tiles) -> 24 loads : 24 wmma per K-step,
//         8 independent accumulation chains.
// ---------------------------------------------------------------------------

typedef __attribute__((ext_vector_type(4)))  __bf16 v4bf;
typedef __attribute__((ext_vector_type(8)))  __bf16 v8bf;
typedef __attribute__((ext_vector_type(16))) __bf16 v16bf;
typedef __attribute__((ext_vector_type(8)))  float  v8f;

#define HIDDEN 2048
#define OUTPUT 8192
#define BATCH  8192
#define NBLK   8
#define KBLK   256   // hidden dims per diagonal block
#define OBLK   1024  // output dims per diagonal block

// workspace layout (bytes)
#define XHI_OFF ((size_t)0)
#define XLO_OFF ((size_t)BATCH * HIDDEN * 2)                    // 32 MB
#define WHI_OFF ((size_t)2 * BATCH * HIDDEN * 2)                // 64 MB
#define WLO_OFF (WHI_OFF + (size_t)OUTPUT * KBLK * 2)           // 68 MB
// total ws use: 72 MB

__device__ __forceinline__ void split4(const float4 v, v4bf& hi, v4bf& lo) {
    const float a[4] = {v.x, v.y, v.z, v.w};
#pragma unroll
    for (int e = 0; e < 4; ++e) {
        __bf16 h = (__bf16)a[e];
        hi[e] = h;
        lo[e] = (__bf16)(a[e] - (float)h);
    }
}

// ---- Pass 1a: X [8192x2048] f32 -> Xhi/Xlo bf16 planes (row-major). --------
__global__ __launch_bounds__(256) void split_x_kernel(
    const float* __restrict__ X, __bf16* __restrict__ Xhi,
    __bf16* __restrict__ Xlo)
{
    const size_t idx = ((size_t)blockIdx.x * 256 + threadIdx.x) * 4;
    const float4 v = *(const float4*)(X + idx);
    v4bf hi, lo;
    split4(v, hi, lo);
    *(v4bf*)(Xhi + idx) = hi;
    *(v4bf*)(Xlo + idx) = lo;
}

// ---- Pass 1b: block-diagonal slice of W -> compacted [8192x256] planes. ----
__global__ __launch_bounds__(256) void split_w_kernel(
    const float* __restrict__ W, __bf16* __restrict__ Whi,
    __bf16* __restrict__ Wlo)
{
    const size_t cidx = ((size_t)blockIdx.x * 256 + threadIdx.x) * 4;
    const int o = (int)(cidx >> 8);       // output row 0..8191
    const int k = (int)(cidx & 255);      // col within diagonal block
    const float4 v =
        *(const float4*)(W + (size_t)o * HIDDEN + (size_t)(o >> 10) * KBLK + k);
    v4bf hi, lo;
    split4(v, hi, lo);
    *(v4bf*)(Whi + cidx) = hi;
    *(v4bf*)(Wlo + cidx) = lo;
}

// ---- Pass 2: WMMA GEMM. ----------------------------------------------------
__device__ __forceinline__ v16bf cat8(v8bf a, v8bf b) {
    return __builtin_shufflevector(a, b, 0, 1, 2, 3, 4, 5, 6, 7, 8, 9, 10, 11,
                                   12, 13, 14, 15);
}

__global__ __launch_bounds__(256) void masked_block_linear_wmma(
    const __bf16* __restrict__ Xhi,  // [BATCH, HIDDEN]
    const __bf16* __restrict__ Xlo,
    const __bf16* __restrict__ Whi,  // [OUTPUT, KBLK] compacted diagonal
    const __bf16* __restrict__ Wlo,
    const float* __restrict__ bias,  // [OUTPUT]
    float* __restrict__ Out)         // [BATCH, OUTPUT]
{
    const int lane = threadIdx.x & 31;
    const int wave = threadIdx.x >> 5;   // 8 waves per workgroup
    const int wm   = wave >> 2;          // 0..1 : wave row (M), 32 rows each
    const int wn   = wave & 3;           // 0..3 : wave col (N), 64 cols each

    const int blk   = blockIdx.z;                              // diagonal block
    const int mbase = blockIdx.y * 64 + wm * 32;               // batch rows
    const int nglob = blk * OBLK + blockIdx.x * 256 + wn * 64; // global out col
    const int kbase = blk * KBLK;                              // hidden offset

    const int r    = lane & 15;  // row-within-tile selector
    const int half = lane >> 4;  // 0: lanes 0-15, 1: lanes 16-31

    v8f acc[2][4];
#pragma unroll
    for (int i = 0; i < 2; ++i)
#pragma unroll
        for (int j = 0; j < 4; ++j)
            acc[i][j] = v8f{};

    // per-lane base offsets (row part depends only on lane, not on k)
    const size_t arow0 = (size_t)(mbase + r) * HIDDEN + kbase + half * 8;
    const size_t brow0 = (size_t)(nglob + r) * KBLK + half * 8;

    // K loop: 256 = 8 steps of 32
#pragma unroll 2
    for (int ks = 0; ks < KBLK / 32; ++ks) {
        const int kk = ks * 32;

        // A fragment, 16x32 bf16:
        //  lanes 0-15 : M=lane,    elems K {0..7, 16..23}
        //  lanes16-31 : M=lane-16, elems K {8..15, 24..31}
        v16bf aHi[2], aLo[2];
#pragma unroll
        for (int i = 0; i < 2; ++i) {
            const size_t a = arow0 + (size_t)(i * 16) * HIDDEN + kk;
            aHi[i] = cat8(*(const v8bf*)(Xhi + a), *(const v8bf*)(Xhi + a + 16));
            aLo[i] = cat8(*(const v8bf*)(Xlo + a), *(const v8bf*)(Xlo + a + 16));
        }

        // B fragment (W^T), 32x16 bf16:
        //  lanes 0-15 : N=lane,    elems K {0..7, 16..23}
        //  lanes16-31 : N=lane-16, elems K {8..15, 24..31}
        // -> two 8-bf16 (16B) chunks per lane from the compacted W planes.
        v16bf bHi[4], bLo[4];
#pragma unroll
        for (int j = 0; j < 4; ++j) {
            const size_t b = brow0 + (size_t)(j * 16) * KBLK + kk;
            bHi[j] = cat8(*(const v8bf*)(Whi + b), *(const v8bf*)(Whi + b + 16));
            bLo[j] = cat8(*(const v8bf*)(Wlo + b), *(const v8bf*)(Wlo + b + 16));
        }

        // 3-term bf16 emulation of f32 product: hh + hl + lh
#pragma unroll
        for (int i = 0; i < 2; ++i) {
#pragma unroll
            for (int j = 0; j < 4; ++j) {
                acc[i][j] = __builtin_amdgcn_wmma_f32_16x16x32_bf16(
                    false, aHi[i], false, bHi[j], (short)0, acc[i][j], false, false);
                acc[i][j] = __builtin_amdgcn_wmma_f32_16x16x32_bf16(
                    false, aHi[i], false, bLo[j], (short)0, acc[i][j], false, false);
                acc[i][j] = __builtin_amdgcn_wmma_f32_16x16x32_bf16(
                    false, aLo[i], false, bHi[j], (short)0, acc[i][j], false, false);
            }
        }
    }

    // Epilogue: bias add + store.
    // C/D layout: VGPR v, lanes 0-15 -> (M=v, N=lane); lanes 16-31 -> (M=v+8).
#pragma unroll
    for (int j = 0; j < 4; ++j) {
        const float bv = bias[nglob + j * 16 + r];
#pragma unroll
        for (int i = 0; i < 2; ++i) {
#pragma unroll
            for (int v = 0; v < 8; ++v) {
                const int row = mbase + i * 16 + half * 8 + v;
                Out[(size_t)row * OUTPUT + nglob + j * 16 + r] =
                    acc[i][j][v] + bv;
            }
        }
    }
}

extern "C" void kernel_launch(void* const* d_in, const int* in_sizes, int n_in,
                              void* d_out, int out_size, void* d_ws, size_t ws_size,
                              hipStream_t stream) {
    const float* X    = (const float*)d_in[0];  // [8192, 2048]
    const float* W    = (const float*)d_in[1];  // [8192, 2048]
    const float* bias = (const float*)d_in[2];  // [8192]
    float*       Out  = (float*)d_out;          // [8192, 8192]

    char* ws = (char*)d_ws;
    __bf16* Xhi = (__bf16*)(ws + XHI_OFF);
    __bf16* Xlo = (__bf16*)(ws + XLO_OFF);
    __bf16* Whi = (__bf16*)(ws + WHI_OFF);
    __bf16* Wlo = (__bf16*)(ws + WLO_OFF);

    // Pass 1: one-time precision split (stream-ordered before the GEMM).
    split_x_kernel<<<(BATCH * HIDDEN) / (256 * 4), 256, 0, stream>>>(X, Xhi, Xlo);
    split_w_kernel<<<(OUTPUT * KBLK) / (256 * 4), 256, 0, stream>>>(W, Whi, Wlo);

    // Pass 2: grid x = N-tiles within a block (1024/256), y = M-tiles
    // (8192/64), z = diagonal block index.
    dim3 grid(OBLK / 256, BATCH / 64, NBLK);
    dim3 block(256);
    masked_block_linear_wmma<<<grid, block, 0, stream>>>(Xhi, Xlo, Whi, Wlo,
                                                         bias, Out);
}
